// MambaModel_11613591568541
// MI455X (gfx1250) — matrix operand order repaired and checked by reading, not compiled, not run on hardware
//
#include <hip/hip_runtime.h>
#include <hip/hip_bf16.h>
#include <math.h>

// ---------------- model constants ----------------
#define BB 2
#define LL 1024
#define DM 768
#define DI 1536
#define DS 16
#define DTR 48
#define NL 8
#define MM (BB * LL)          // 2048 token rows
#define N_XZ (2 * DI)         // 3072
#define N_DBL_PAD 128         // 80 -> padded
#define K_DT_PAD 64           // 48 -> padded
#define KT 64                 // GEMM k-slab staged in LDS

typedef __attribute__((ext_vector_type(16))) __bf16 v16bf;
typedef __attribute__((ext_vector_type(8))) float v8f;
typedef int v4i __attribute__((ext_vector_type(4)));

#if defined(__has_builtin)
#if __has_builtin(__builtin_amdgcn_global_load_async_to_lds_b128)
#define HAVE_ASYNC_LDS 1
#endif
#endif

__device__ __forceinline__ unsigned short f2bf(float f) {
    unsigned int x = __float_as_uint(f);
    unsigned int r = x + 0x7fffu + ((x >> 16) & 1u);   // round-to-nearest-even
    return (unsigned short)(r >> 16);
}

union FragU { uint4 q[2]; v16bf v; };

// A fragment (16x32 bf16 slab, row-major, stride KT):
// lane<16: row=lane, K = 0..15 ; lane>=16: row=lane-16, K = 16..31
__device__ __forceinline__ v16bf load_fragA(const unsigned short* p) {
    FragU f;
    f.q[0] = *(const uint4*)(p);
    f.q[1] = *(const uint4*)(p + 8);
    return f.v;
}
// B fragment (32x16 bf16 from W[N,K] layout, column n contiguous in K):
// lane<16: col=lane, K=0..15 ; lane>=16: col=lane-16, K=16..31
__device__ __forceinline__ v16bf load_fragB(const unsigned short* p) {
    FragU f;
    f.q[0] = *(const uint4*)(p);
    f.q[1] = *(const uint4*)(p + 8);
    return f.v;
}

// 16-byte global -> LDS copy, async (ASYNCcnt) when toolchain supports it
__device__ __forceinline__ void cp_async16(unsigned short* lds, const unsigned short* g) {
#ifdef HAVE_ASYNC_LDS
    __builtin_amdgcn_global_load_async_to_lds_b128(
        (__attribute__((address_space(1))) v4i*)g,
        (__attribute__((address_space(3))) v4i*)lds, 0, 0);
#else
    *(uint4*)lds = *(const uint4*)g;
#endif
}

__device__ __forceinline__ void wait_cp_async() {
#ifdef HAVE_ASYNC_LDS
#if __has_builtin(__builtin_amdgcn_s_wait_asynccnt)
    __builtin_amdgcn_s_wait_asynccnt(0);
#else
    asm volatile("s_wait_asynccnt 0x0" ::: "memory");
#endif
#endif
}

// C/D 16x16 f32: vgpr v -> M = (lane>=16 ? 8 : 0) + v, N = lane&15
__device__ __forceinline__ void store_tile(v8f c, float* __restrict__ C, int ldc,
                                           int r0, int c0, int lane,
                                           const float* __restrict__ bias,
                                           const float* __restrict__ resid, int act) {
    int n  = c0 + (lane & 15);
    int m0 = r0 + (lane >> 4) * 8;
    float bn = (act == 1) ? bias[n] : 0.f;
#pragma unroll
    for (int v = 0; v < 8; ++v) {
        size_t idx = (size_t)(m0 + v) * ldc + n;
        float val = c[v];
        if (act == 1) {                       // softplus(x + b_dt[n])
            float x = val + bn;
            val = (x > 20.f) ? x : log1pf(__expf(x));
        } else if (act == 2) {                // residual accumulate
            val += resid[idx];
        }
        C[idx] = val;
    }
}

// ---------------- WMMA GEMM: C[M,N] = A[M,K](bf16,rm) x W[N,K](bf16,rm)^T ----------------
// block = 256 threads = 8 waves (4x2), block tile 128x64.
// Per 64-wide k-slab: A(128x64) and B(64x64) staged in LDS via async copies,
// then each wave runs two 16x16x32 WMMA k-steps per accumulator from LDS.
// Requires M%128==0, N%64==0, K%64==0 (buffers padded accordingly).
__global__ __launch_bounds__(256) void k_gemm_bf16(
    const unsigned short* __restrict__ A, const unsigned short* __restrict__ W,
    float* __restrict__ C, const float* __restrict__ bias,
    const float* __restrict__ resid, int M, int N, int K, int act) {
    __shared__ unsigned short sA[128 * KT];   // [row][k]
    __shared__ unsigned short sB[64 * KT];    // [col][k]
    int tid  = threadIdx.x;
    int lane = tid & 31;
    int wid  = tid >> 5;
    int rowBlk = blockIdx.y * 128;
    int colBlk = blockIdx.x * 64;
    int wrow = (wid >> 1) * 32;
    int wcol = (wid & 1) * 32;
    int r    = lane & 15;
    int half = lane >> 4;

    // per-lane LDS fragment pointers (stride KT)
    const unsigned short* la0 = sA + (wrow + r) * KT + half * 16;
    const unsigned short* la1 = la0 + 16 * KT;
    const unsigned short* lb0 = sB + (wcol + r) * KT + half * 16;
    const unsigned short* lb1 = lb0 + 16 * KT;

    v8f c00 = {}, c01 = {}, c10 = {}, c11 = {};
    for (int k0 = 0; k0 < K; k0 += KT) {
        // stage A tile: 128 rows x 64 k = 1024 x 16B chunks, 4 per thread
#pragma unroll
        for (int i = 0; i < 4; ++i) {
            int chunk = tid + i * 256;
            int arow = chunk >> 3;
            int ak   = (chunk & 7) * 8;
            cp_async16(&sA[arow * KT + ak],
                       A + (size_t)(rowBlk + arow) * K + k0 + ak);
        }
        // stage B tile: 64 rows x 64 k = 512 x 16B chunks, 2 per thread
#pragma unroll
        for (int i = 0; i < 2; ++i) {
            int chunk = tid + i * 256;
            int brow = chunk >> 3;
            int bk   = (chunk & 7) * 8;
            cp_async16(&sB[brow * KT + bk],
                       W + (size_t)(colBlk + brow) * K + k0 + bk);
        }
        wait_cp_async();
        __syncthreads();
#pragma unroll
        for (int kk = 0; kk < KT; kk += 32) {
            v16bf a0 = load_fragA(la0 + kk);
            v16bf a1 = load_fragA(la1 + kk);
            v16bf b0 = load_fragB(lb0 + kk);
            v16bf b1 = load_fragB(lb1 + kk);
            c00 = __builtin_amdgcn_wmma_f32_16x16x32_bf16(false, a0, false, b0, (short)0, c00, false, false);
            c01 = __builtin_amdgcn_wmma_f32_16x16x32_bf16(false, a0, false, b1, (short)0, c01, false, false);
            c10 = __builtin_amdgcn_wmma_f32_16x16x32_bf16(false, a1, false, b0, (short)0, c10, false, false);
            c11 = __builtin_amdgcn_wmma_f32_16x16x32_bf16(false, a1, false, b1, (short)0, c11, false, false);
        }
        __syncthreads();
    }
    int row0 = rowBlk + wrow, col0 = colBlk + wcol;
    store_tile(c00, C, N, row0,      col0,      lane, bias, resid, act);
    store_tile(c01, C, N, row0,      col0 + 16, lane, bias, resid, act);
    store_tile(c10, C, N, row0 + 16, col0,      lane, bias, resid, act);
    store_tile(c11, C, N, row0 + 16, col0 + 16, lane, bias, resid, act);
}

// ---------------- misc kernels ----------------
__global__ __launch_bounds__(256) void k_transpose_in(const float* __restrict__ batch,
                                                      float* __restrict__ h) {
    int idx = blockIdx.x * 256 + threadIdx.x;          // over B*L*D
    int d = idx % DM;
    int m = idx / DM;
    int b = m / LL, l = m % LL;
    h[idx] = batch[((size_t)b * DM + d) * LL + l];
}

// fp32 -> bf16 weight conversion with zero padding, across all layers
__global__ __launch_bounds__(256) void k_cvt_pad(const float* __restrict__ src,
                                                 unsigned short* __restrict__ dst,
                                                 int N, int K, int Npad, int Kpad, int total) {
    int idx = blockIdx.x * 256 + threadIdx.x;
    if (idx >= total) return;
    int k = idx % Kpad;
    int t = idx / Kpad;
    int n = t % Npad;
    int lyr = t / Npad;
    float v = (n < N && k < K) ? src[((size_t)lyr * N + n) * K + k] : 0.f;
    dst[idx] = f2bf(v);
}

__global__ __launch_bounds__(256) void k_rmsnorm_bf16(const float* __restrict__ x,
                                                      const float* __restrict__ w,
                                                      unsigned short* __restrict__ out) {
    __shared__ float red[256];
    int row = blockIdx.x, tid = threadIdx.x;
    const float* xr = x + (size_t)row * DM;
    float p = 0.f;
    for (int d = tid; d < DM; d += 256) { float v = xr[d]; p += v * v; }
    red[tid] = p; __syncthreads();
    for (int o = 128; o > 0; o >>= 1) { if (tid < o) red[tid] += red[tid + o]; __syncthreads(); }
    float scale = rsqrtf(red[0] / DM + 1e-5f);
    for (int d = tid; d < DM; d += 256)
        out[(size_t)row * DM + d] = f2bf(xr[d] * w[d] * scale);
}

// causal depthwise conv1d (k=4) + bias + SiLU ; input = xz[:, :DI]
__global__ __launch_bounds__(256) void k_conv_silu(const float* __restrict__ xz,
                                                   const float* __restrict__ cw,
                                                   const float* __restrict__ cb,
                                                   float* __restrict__ u,
                                                   unsigned short* __restrict__ ubf) {
    int idx = blockIdx.x * 256 + threadIdx.x;          // over M*DI
    int c = idx % DI;
    int m = idx / DI;
    int b = m / LL, l = m % LL;
    float acc = cb[c];
#pragma unroll
    for (int j = 0; j < 4; ++j) {
        int t = l - 3 + j;
        if (t >= 0) acc += cw[c * 4 + j] * xz[((size_t)b * LL + t) * N_XZ + c];
    }
    float s = acc / (1.f + __expf(-acc));              // SiLU
    u[idx] = s;
    ubf[idx] = f2bf(s);
}

// extract dt-projection input (cols 0..47 of dbl), pad K to 64, to bf16
__global__ __launch_bounds__(256) void k_dtp(const float* __restrict__ dbl,
                                             unsigned short* __restrict__ dtp) {
    int idx = blockIdx.x * 256 + threadIdx.x;          // over M*64
    int r = idx % K_DT_PAD;
    int m = idx / K_DT_PAD;
    float v = (r < DTR) ? dbl[(size_t)m * N_DBL_PAD + r] : 0.f;
    dtp[idx] = f2bf(v);
}

// sequential selective scan: one thread per (b, channel); B_t/C_t staged in LDS
__global__ __launch_bounds__(256) void k_scan(const float* __restrict__ delta,
                                              const float* __restrict__ u,
                                              const float* __restrict__ dbl,
                                              const float* __restrict__ A_log,
                                              const float* __restrict__ Dsk,
                                              float* __restrict__ y) {
    __shared__ float sBC[32];
    int b = blockIdx.x / (DI / 256);
    int c = (blockIdx.x % (DI / 256)) * 256 + threadIdx.x;
    float a[DS], hs[DS];
#pragma unroll
    for (int s = 0; s < DS; ++s) { a[s] = -__expf(A_log[c * DS + s]); hs[s] = 0.f; }
    float dsk = Dsk[c];
    for (int l = 0; l < LL; ++l) {
        int row = b * LL + l;
        if (threadIdx.x < 32) sBC[threadIdx.x] = dbl[(size_t)row * N_DBL_PAD + DTR + threadIdx.x];
        __syncthreads();
        float dt = delta[(size_t)row * DI + c];
        float ut = u[(size_t)row * DI + c];
        float du = dt * ut;
        float acc = 0.f;
#pragma unroll
        for (int s = 0; s < DS; ++s) {
            hs[s] = __expf(dt * a[s]) * hs[s] + du * sBC[s];
            acc += hs[s] * sBC[DS + s];
        }
        y[(size_t)row * DI + c] = acc + ut * dsk;
        __syncthreads();
    }
}

// gate: ybf = bf16(y * silu(z)), z = xz[:, DI + c]
__global__ __launch_bounds__(256) void k_zmul(const float* __restrict__ xz,
                                              const float* __restrict__ y,
                                              unsigned short* __restrict__ ybf) {
    int idx = blockIdx.x * 256 + threadIdx.x;          // over M*DI
    int c = idx % DI;
    int m = idx / DI;
    float z = xz[(size_t)m * N_XZ + DI + c];
    float zv = z / (1.f + __expf(-z));
    ybf[idx] = f2bf(y[idx] * zv);
}

// final: pick last valid token per batch (from mask), rmsnorm with normf_w
__global__ __launch_bounds__(256) void k_final(const float* __restrict__ h,
                                               const unsigned char* __restrict__ mask,
                                               const float* __restrict__ w,
                                               float* __restrict__ out) {
    __shared__ float red[256];
    __shared__ int sIdx;
    int b = blockIdx.x, tid = threadIdx.x;
    if (tid == 0) {
        int idx = LL;
        for (int l = 0; l < LL; ++l)
            if (!mask[b * LL + l]) { idx = l; break; }
        sIdx = idx - 1;
    }
    __syncthreads();
    const float* row = h + ((size_t)b * LL + sIdx) * DM;
    float p = 0.f;
    for (int d = tid; d < DM; d += 256) p += row[d] * row[d];
    red[tid] = p; __syncthreads();
    for (int o = 128; o > 0; o >>= 1) { if (tid < o) red[tid] += red[tid + o]; __syncthreads(); }
    float scale = rsqrtf(red[0] / DM + 1e-5f);
    for (int d = tid; d < DM; d += 256) out[b * DM + d] = row[d] * w[d] * scale;
}

// ---------------- host launch ----------------
extern "C" void kernel_launch(void* const* d_in, const int* in_sizes, int n_in,
                              void* d_out, int out_size, void* d_ws, size_t ws_size,
                              hipStream_t stream) {
    const float* batch          = (const float*)d_in[0];
    const unsigned char* maskp  = (const unsigned char*)d_in[1];   // jnp.bool_ -> 1 byte
    const float* norm_w         = (const float*)d_in[2];
    const float* W_in           = (const float*)d_in[3];
    const float* conv_w         = (const float*)d_in[4];
    const float* conv_b         = (const float*)d_in[5];
    const float* W_x            = (const float*)d_in[6];
    const float* W_dt           = (const float*)d_in[7];
    const float* b_dt           = (const float*)d_in[8];
    const float* A_log          = (const float*)d_in[9];
    const float* D_skip         = (const float*)d_in[10];
    const float* W_out          = (const float*)d_in[11];
    const float* normf_w        = (const float*)d_in[12];
    float* out                  = (float*)d_out;

    char* cur = (char*)d_ws;
    auto alloc = [&](size_t bytes) -> void* {
        void* p = cur;
        cur += (bytes + 255) & ~(size_t)255;
        return p;
    };
    float*          h       = (float*)alloc((size_t)MM * DM * 4);
    unsigned short* xn_bf   = (unsigned short*)alloc((size_t)MM * DM * 2);
    float*          xz      = (float*)alloc((size_t)MM * N_XZ * 4);
    float*          u       = (float*)alloc((size_t)MM * DI * 4);
    unsigned short* u_bf    = (unsigned short*)alloc((size_t)MM * DI * 2);
    float*          dbl     = (float*)alloc((size_t)MM * N_DBL_PAD * 4);
    unsigned short* dtp_bf  = (unsigned short*)alloc((size_t)MM * K_DT_PAD * 2);
    float*          delta   = (float*)alloc((size_t)MM * DI * 4);
    float*          y       = (float*)alloc((size_t)MM * DI * 4);
    unsigned short* y_bf    = (unsigned short*)alloc((size_t)MM * DI * 2);
    unsigned short* Win_bf  = (unsigned short*)alloc((size_t)NL * N_XZ * DM * 2);
    unsigned short* Wx_bf   = (unsigned short*)alloc((size_t)NL * N_DBL_PAD * DI * 2);
    unsigned short* Wdt_bf  = (unsigned short*)alloc((size_t)NL * DI * K_DT_PAD * 2);
    unsigned short* Wout_bf = (unsigned short*)alloc((size_t)NL * DM * DI * 2);

    // h = transpose(batch)
    k_transpose_in<<<(MM * DM) / 256, 256, 0, stream>>>(batch, h);

    // weights -> bf16 (padded) once per launch (deterministic)
    {
        int t;
        t = NL * N_XZ * DM;
        k_cvt_pad<<<(t + 255) / 256, 256, 0, stream>>>(W_in, Win_bf, N_XZ, DM, N_XZ, DM, t);
        t = NL * N_DBL_PAD * DI;
        k_cvt_pad<<<(t + 255) / 256, 256, 0, stream>>>(W_x, Wx_bf, DTR + 2 * DS, DI, N_DBL_PAD, DI, t);
        t = NL * DI * K_DT_PAD;
        k_cvt_pad<<<(t + 255) / 256, 256, 0, stream>>>(W_dt, Wdt_bf, DI, DTR, DI, K_DT_PAD, t);
        t = NL * DM * DI;
        k_cvt_pad<<<(t + 255) / 256, 256, 0, stream>>>(W_out, Wout_bf, DM, DI, DM, DI, t);
    }

    for (int i = 0; i < NL; ++i) {
        // 1. pre-norm -> bf16 activations
        k_rmsnorm_bf16<<<MM, 256, 0, stream>>>(h, norm_w + (size_t)i * DM, xn_bf);

        // 2. xz = xn @ W_in^T   [2048 x 3072], K=768
        k_gemm_bf16<<<dim3(N_XZ / 64, MM / 128), 256, 0, stream>>>(
            xn_bf, Win_bf + (size_t)i * N_XZ * DM, xz, nullptr, nullptr, MM, N_XZ, DM, 0);

        // 3. u = silu(causal depthwise conv(xc) + cb)
        k_conv_silu<<<(MM * DI) / 256, 256, 0, stream>>>(
            xz, conv_w + (size_t)i * DI * 4, conv_b + (size_t)i * DI, u, u_bf);

        // 4. dbl = u @ W_x^T   [2048 x 128(pad of 80)], K=1536
        k_gemm_bf16<<<dim3(N_DBL_PAD / 64, MM / 128), 256, 0, stream>>>(
            u_bf, Wx_bf + (size_t)i * N_DBL_PAD * DI, dbl, nullptr, nullptr, MM, N_DBL_PAD, DI, 0);

        // 5. dtp (cols 0..47, K-padded to 64) -> bf16
        k_dtp<<<(MM * K_DT_PAD) / 256, 256, 0, stream>>>(dbl, dtp_bf);

        // 6. delta = softplus(dtp @ W_dt^T + b_dt)   [2048 x 1536], K=64
        k_gemm_bf16<<<dim3(DI / 64, MM / 128), 256, 0, stream>>>(
            dtp_bf, Wdt_bf + (size_t)i * DI * K_DT_PAD, delta,
            b_dt + (size_t)i * DI, nullptr, MM, DI, K_DT_PAD, 1);

        // 7. selective scan (+ u * D_skip)
        k_scan<<<BB * (DI / 256), 256, 0, stream>>>(
            delta, u, dbl, A_log + (size_t)i * DI * DS, D_skip + (size_t)i * DI, y);

        // 8. gate with silu(z) -> bf16
        k_zmul<<<(MM * DI) / 256, 256, 0, stream>>>(xz, y, y_bf);

        // 9. h += ymod @ W_out^T   [2048 x 768], K=1536 (fused residual)
        k_gemm_bf16<<<dim3(DM / 64, MM / 128), 256, 0, stream>>>(
            y_bf, Wout_bf + (size_t)i * DM * DI, h, nullptr, h, MM, DM, DI, 2);
    }

    // final: last valid token per batch + rmsnorm
    k_final<<<BB, 256, 0, stream>>>(h, maskp, normf_w, out);

    (void)in_sizes; (void)n_in; (void)out_size; (void)ws_size;
}
